// FusionAttention_76527727280508
// MI455X (gfx1250) — matrix operand order repaired and checked by reading, not compiled, not run on hardware
//
#include <hip/hip_runtime.h>

typedef __attribute__((ext_vector_type(16))) _Float16 v16h;
typedef __attribute__((ext_vector_type(8)))  _Float16 v8h;
typedef __attribute__((ext_vector_type(8)))  float    v8f;

#define WMMA_F16(a, b, c) \
  __builtin_amdgcn_wmma_f32_16x16x32_f16(false, (a), false, (b), (short)0, (c), false, false)

static constexpr int HWp  = 128 * 128;   // 16384 pixels per image
static constexpr int NPIX = 4 * HWp;     // 65536 pixels total

// ---- CDNA5 async global->LDS copy (ASYNCcnt) --------------------------------
__device__ __forceinline__ unsigned lds_off(const void* p) {
  return (unsigned)(size_t)p;   // low 32 bits of generic LDS address
}
__device__ __forceinline__ void async_ld16(unsigned lds, const void* g) {
  asm volatile("global_load_async_to_lds_b128 %0, %1, off"
               :: "v"(lds), "v"((unsigned long long)(size_t)g) : "memory");
}
__device__ __forceinline__ void wait_async0() {
  asm volatile("s_wait_asynccnt 0x0" ::: "memory");
}

// ---------------------------------------------------------------------------
// Transpose-convert NCHW f32 -> [pixel][channel] f16 (tile 32c x 128p via LDS).
// Used for x, y, and the depthwise-conv output feeding the final GEMM.
// ---------------------------------------------------------------------------
__global__ __launch_bounds__(256)
void k_cvtT(const float* __restrict__ src, _Float16* __restrict__ dstT) {
  __shared__ _Float16 T[128][32];
  const int tid = threadIdx.x;
  const int p0 = blockIdx.x * 128;
  const int c0 = blockIdx.y * 32;
  const int b = p0 >> 14, hw0 = p0 & (HWp - 1);
  {
    const int c  = tid >> 3;
    const int pb = (tid & 7) * 16;
    const float* s = src + ((size_t)(b * 256 + c0 + c)) * HWp + hw0 + pb;
#pragma unroll
    for (int j = 0; j < 16; ++j) T[pb + j][c] = (_Float16)s[j];
  }
  __syncthreads();
  {
    const int row = tid >> 1;
    const int hb  = (tid & 1) * 16;
    _Float16* d = dstT + (size_t)(p0 + row) * 256 + c0 + hb;
    v8h a = *(const v8h*)&T[row][hb];
    v8h b2 = *(const v8h*)&T[row][hb + 8];
    *(v8h*)d = a;
    *(v8h*)(d + 8) = b2;
  }
}

// ---------------------------------------------------------------------------
// Weights -> f16, BN scales folded.
//   qkvwh[oc][c] (768x256), pwwh[oc][c] (256x256)
//   w1h[tap][oc][c], tap 0..8 = s1*conv3x3 taps, tap 9 = s2*conv1x1
// ---------------------------------------------------------------------------
__global__ __launch_bounds__(256)
void k_cvtw(const float* __restrict__ qkvw,
            const float* __restrict__ w1, const float* __restrict__ s1,
            const float* __restrict__ w2, const float* __restrict__ s2,
            const float* __restrict__ pww,
            _Float16* __restrict__ qkvwh, _Float16* __restrict__ w1h,
            _Float16* __restrict__ pwwh) {
  const int i = blockIdx.x * 256 + threadIdx.x;
  if (i < 768 * 256) qkvwh[i] = (_Float16)qkvw[i];
  if (i < 256 * 256) pwwh[i] = (_Float16)pww[i];
  if (i < 10 * 256 * 256) {
    const int tap = i >> 16;
    const int oc  = (i >> 8) & 255;
    const int c   = i & 255;
    float v;
    if (tap < 9) v = s1[oc] * w1[((size_t)oc * 256 + c) * 9 + tap];
    else         v = s2[oc] * w2[(size_t)oc * 256 + c];
    w1h[i] = (_Float16)v;
  }
}

// ---------------------------------------------------------------------------
// qkv 1x1 conv GEMM.  xT f16 [p][256] x wh f16 [768][256] -> outT f16 [p][768]
// Tiles filled purely by async global->LDS b128.
// ---------------------------------------------------------------------------
__global__ __launch_bounds__(256)
void k_qkv(const _Float16* __restrict__ xT, const _Float16* __restrict__ wh,
           _Float16* __restrict__ outT) {
  __shared__ _Float16 Xs[128][32];   // [m][k], rows 64B
  __shared__ _Float16 Ws[64][32];    // [n][k], rows 64B

  const int tid  = threadIdx.x;
  const int lane = tid & 31, wid = tid >> 5;
  const int waveM = (wid & 3) * 32, waveN = (wid >> 2) * 32;
  const int m0 = blockIdx.x * 128;
  const int n0 = blockIdx.y * 64;
  const int hi = lane >> 4, ln = lane & 15;

  v8f acc[2][2] = {};

  for (int k0 = 0; k0 < 256; k0 += 32) {
#pragma unroll
    for (int cc = 0; cc < 2; ++cc) {
      const int chunk = tid + cc * 256;
      const int row = chunk >> 2, kc = (chunk & 3) * 8;
      async_ld16(lds_off(&Xs[row][kc]),
                 xT + (size_t)(m0 + row) * 256 + k0 + kc);
    }
    {
      const int row = tid >> 2, kc = (tid & 3) * 8;
      async_ld16(lds_off(&Ws[row][kc]),
                 wh + (size_t)(n0 + row) * 256 + k0 + kc);
    }
    if (k0 + 32 < 256)
      __builtin_prefetch(xT + (size_t)(m0 + (tid >> 1)) * 256 + k0 + 32, 0, 1);
    wait_async0();
    __syncthreads();

#pragma unroll
    for (int ni = 0; ni < 2; ++ni) {
      const int col = waveN + ni * 16 + ln;
      v16h bf = *(const v16h*)&Ws[col][hi * 16];          // K = 16*hi + h
#pragma unroll
      for (int mi = 0; mi < 2; ++mi) {
        const int row = waveM + mi * 16 + ln;
        v8h lo = *(const v8h*)&Xs[row][hi * 16];          // K = 16*hi + h
        v8h up = *(const v8h*)&Xs[row][hi * 16 + 8];      // K = 16*hi + 8 + h
        v16h af;
#pragma unroll
        for (int j = 0; j < 8; ++j) { af[j] = lo[j]; af[8 + j] = up[j]; }
        acc[mi][ni] = WMMA_F16(af, bf, acc[mi][ni]);
      }
    }
    __syncthreads();
  }

#pragma unroll
  for (int mi = 0; mi < 2; ++mi)
#pragma unroll
    for (int ni = 0; ni < 2; ++ni) {
      const int n = n0 + waveN + ni * 16 + ln;
#pragma unroll
      for (int r = 0; r < 8; ++r) {
        const int m = m0 + waveM + mi * 16 + r + 8 * hi;
        outT[(size_t)m * 768 + n] = (_Float16)acc[mi][ni][r];
      }
    }
}

// ---------------------------------------------------------------------------
// Local branch implicit GEMM over 10 taps (9x conv3x3 + 1x conv1x1),
// BN scales pre-folded; biases added in epilogue. M-tile = 1 image row.
// ---------------------------------------------------------------------------
__global__ __launch_bounds__(256)
void k_local(const _Float16* __restrict__ yT, const _Float16* __restrict__ w1h,
             const float* __restrict__ b1, const float* __restrict__ b2,
             float* __restrict__ localo) {
  __shared__ _Float16 Xs[128][32];
  __shared__ _Float16 Ws[64][32];

  const int tid  = threadIdx.x;
  const int lane = tid & 31, wid = tid >> 5;
  const int waveM = (wid & 3) * 32, waveN = (wid >> 2) * 32;
  const int b = blockIdx.x >> 7, h = blockIdx.x & 127;
  const int n0 = blockIdx.y * 64;
  const int hi = lane >> 4, ln = lane & 15;

  v8f acc[2][2] = {};

  for (int tap = 0; tap < 10; ++tap) {
    const int dy = (tap < 9) ? tap / 3 - 1 : 0;
    const int dx = (tap < 9) ? tap % 3 - 1 : 0;
    const int sr = h + dy;
    if ((unsigned)sr >= 128u) continue;   // zero rows contribute nothing
    const size_t prow = (size_t)(b * HWp + sr * 128);

    for (int k0 = 0; k0 < 256; k0 += 32) {
#pragma unroll
      for (int cc = 0; cc < 2; ++cc) {
        const int chunk = tid + cc * 256;
        const int row = chunk >> 2, kc = (chunk & 3) * 8;
        const int sm = row + dx;
        if (sm >= 0 && sm < 128) {
          async_ld16(lds_off(&Xs[row][kc]),
                     yT + (prow + sm) * 256 + k0 + kc);
        } else {
          v8h z = {};
          *(v8h*)&Xs[row][kc] = z;
        }
      }
      {
        const int row = tid >> 2, kc = (tid & 3) * 8;
        async_ld16(lds_off(&Ws[row][kc]),
                   w1h + ((size_t)tap * 256 + n0 + row) * 256 + k0 + kc);
      }
      wait_async0();
      __syncthreads();

#pragma unroll
      for (int ni = 0; ni < 2; ++ni) {
        const int col = waveN + ni * 16 + ln;
        v16h bf = *(const v16h*)&Ws[col][hi * 16];
#pragma unroll
        for (int mi = 0; mi < 2; ++mi) {
          const int row = waveM + mi * 16 + ln;
          v8h lo = *(const v8h*)&Xs[row][hi * 16];
          v8h up = *(const v8h*)&Xs[row][hi * 16 + 8];
          v16h af;
#pragma unroll
          for (int j = 0; j < 8; ++j) { af[j] = lo[j]; af[8 + j] = up[j]; }
          acc[mi][ni] = WMMA_F16(af, bf, acc[mi][ni]);
        }
      }
      __syncthreads();
    }
  }

#pragma unroll
  for (int mi = 0; mi < 2; ++mi)
#pragma unroll
    for (int ni = 0; ni < 2; ++ni) {
      const int oc = n0 + waveN + ni * 16 + ln;
      const float bias = b1[oc] + b2[oc];
#pragma unroll
      for (int r = 0; r < 8; ++r) {
        const int m = waveM + mi * 16 + r + 8 * hi;   // = w coordinate
        localo[((size_t)(b * 256 + oc)) * HWp + h * 128 + m] =
            acc[mi][ni][r] + bias;
      }
    }
}

// ---------------------------------------------------------------------------
// Window attention. One wave per (window, head). V staged through LDS.
// ---------------------------------------------------------------------------
__global__ __launch_bounds__(32)
void k_attn(const _Float16* __restrict__ qkvT,
            const float* __restrict__ rel_bias,
            float* __restrict__ outo) {
  __shared__ float    Sm[64][64];
  __shared__ _Float16 Pm[64][64];
  __shared__ _Float16 Vs[64][16];

  const int lane = threadIdx.x;
  const int head = blockIdx.x & 15;
  const int wi   = blockIdx.x >> 4;
  const int b = wi >> 8, rem = wi & 255, wy = rem >> 4, wx = rem & 15;
  const int hi = lane >> 4, ln = lane & 15;
  const float scale = 0.25f;   // 16^-0.5

  auto pixof = [&](int t) {
    return b * HWp + (wy * 8 + (t >> 3)) * 128 + wx * 8 + (t & 7);
  };

  // stage V (contiguous v16h per token) early to overlap with S compute
#pragma unroll
  for (int tt = 0; tt < 2; ++tt) {
    const int t = lane * 2 + tt;
    *(v16h*)&Vs[t][0] =
        *(const v16h*)(qkvT + (size_t)pixof(t) * 768 + 512 + head * 16);
  }

  // Q fragments (A operand), zero-padded K 16->32
  v16h qa[4];
#pragma unroll
  for (int mi = 0; mi < 4; ++mi) {
    const int t = mi * 16 + ln;
    v16h qv = *(const v16h*)(qkvT + (size_t)pixof(t) * 768 + head * 16);
    v16h a;
#pragma unroll
    for (int hh = 0; hh < 16; ++hh) {
      const int kk = 16 * hi + hh;
      a[hh] = (kk < 16) ? qv[kk] : (_Float16)0.f;
    }
    qa[mi] = a;
  }

  // S = Q K^T * scale + relative position bias -> LDS
#pragma unroll
  for (int ni = 0; ni < 4; ++ni) {
    const int t = ni * 16 + ln;
    v16h bf;
#pragma unroll
    for (int hh = 0; hh < 16; ++hh) bf[hh] = (_Float16)0.f;
    if (hi == 0) {   // lanes 16..31 hold K=16..31 (zero pad)
      bf = *(const v16h*)(qkvT + (size_t)pixof(t) * 768 + 256 + head * 16);
    }
#pragma unroll
    for (int mi = 0; mi < 4; ++mi) {
      v8f z = {};
      v8f s = WMMA_F16(qa[mi], bf, z);
#pragma unroll
      for (int r = 0; r < 8; ++r) {
        const int row = mi * 16 + r + 8 * hi;
        const int col = ni * 16 + ln;
        const int qy = row >> 3, qx = row & 7, ky = col >> 3, kx = col & 7;
        const int ridx = (qy - ky + 7) * 15 + (qx - kx + 7);
        Sm[row][col] = s[r] * scale + rel_bias[ridx * 16 + head];
      }
    }
  }
  __syncthreads();

  // row softmax (2 rows per lane), P (f16) -> LDS
#pragma unroll
  for (int rr = 0; rr < 2; ++rr) {
    const int row = lane + 32 * rr;
    float mx = -1e30f;
#pragma unroll
    for (int c = 0; c < 64; ++c) mx = fmaxf(mx, Sm[row][c]);
    float sum = 0.f;
#pragma unroll
    for (int c = 0; c < 64; ++c) sum += __expf(Sm[row][c] - mx);
    const float inv = 1.f / sum;
#pragma unroll
    for (int c = 0; c < 64; ++c)
      Pm[row][c] = (_Float16)(__expf(Sm[row][c] - mx) * inv);
  }
  __syncthreads();

  // O = P V  (M=64, N=16, K=64 in two K=32 steps), V from LDS
  v8f oacc[4] = {};
#pragma unroll
  for (int ks = 0; ks < 2; ++ks) {
    v16h bf;   // B = V: lane ln = d, half h -> token = 8*hi + h (+8 if h>=8) + 32*ks
#pragma unroll
    for (int hh = 0; hh < 16; ++hh) {
      bf[hh] = Vs[hi * 8 + hh + (hh >= 8 ? 8 : 0) + 32 * ks][ln];
    }
#pragma unroll
    for (int mi = 0; mi < 4; ++mi) {
      const int row = mi * 16 + ln;
      v8h lo = *(const v8h*)&Pm[row][32 * ks + hi * 8];
      v8h up = *(const v8h*)&Pm[row][32 * ks + 16 + hi * 8];
      v16h af;
#pragma unroll
      for (int j = 0; j < 8; ++j) { af[j] = lo[j]; af[8 + j] = up[j]; }
      oacc[mi] = WMMA_F16(af, bf, oacc[mi]);
    }
  }

#pragma unroll
  for (int mi = 0; mi < 4; ++mi)
#pragma unroll
    for (int r = 0; r < 8; ++r) {
      const int t = mi * 16 + r + 8 * hi;
      const int c = head * 16 + ln;
      const int p = pixof(t);
      const int bb = p >> 14, hw = p & (HWp - 1);
      outo[((size_t)(bb * 256 + c)) * HWp + hw] = oacc[mi][r];
    }
}

// ---------------------------------------------------------------------------
// o = avgpool_v(out) + avgpool_h(out) + local
// ---------------------------------------------------------------------------
__global__ __launch_bounds__(256)
void k_pool(const float* __restrict__ att, const float* __restrict__ loc,
            float* __restrict__ o) {
  const size_t idx = (size_t)blockIdx.x * 256 + threadIdx.x;
  const int w = idx & 127, h = (idx >> 7) & 127;
  const size_t bc = idx >> 14;
  const float* base = att + bc * HWp;

  float sx = 0.f, sy = 0.f;
#pragma unroll
  for (int j = -3; j <= 4; ++j) {
    const int r = h + j;
    if (r >= 0 && r <= 128) sx += base[((r == 128) ? 126 : r) * 128 + w];
    const int s = w + j;
    if (s >= 0 && s <= 128) sy += base[h * 128 + ((s == 128) ? 126 : s)];
  }
  o[idx] = (sx + sy) * 0.125f + loc[idx];
}

// ---------------------------------------------------------------------------
// Depthwise 8x8 conv (reflect-pad 1 + zero-pad 3) + BN.
// ---------------------------------------------------------------------------
__global__ __launch_bounds__(256)
void k_dw(const float* __restrict__ o, const float* __restrict__ dww,
          const float* __restrict__ bs, const float* __restrict__ bb,
          float* __restrict__ out) {
  const size_t idx = (size_t)blockIdx.x * 256 + threadIdx.x;
  const int w = idx & 127, h = (idx >> 7) & 127;
  const size_t bc = idx >> 14;
  const int c = (int)(bc & 255);
  const float* base = o + bc * HWp;
  const float* wk = dww + c * 64;

  float acc = 0.f;
#pragma unroll
  for (int i = 0; i < 8; ++i) {
    const int r = h + i - 3;
    if (r < 0 || r > 128) continue;
    const int rr = (r == 128) ? 126 : r;
#pragma unroll
    for (int j = 0; j < 8; ++j) {
      const int s = w + j - 3;
      if (s < 0 || s > 128) continue;
      const int ss = (s == 128) ? 126 : s;
      acc += base[rr * 128 + ss] * wk[i * 8 + j];
    }
  }
  out[idx] = acc * bs[c] + bb[c];
}

// ---------------------------------------------------------------------------
// Projection pointwise GEMM, async f16 feeds (same structure as k_qkv),
// N=256, f32 NCHW epilogue into d_out.
// ---------------------------------------------------------------------------
__global__ __launch_bounds__(256)
void k_pwa(const _Float16* __restrict__ xT, const _Float16* __restrict__ wh,
           float* __restrict__ outp) {
  __shared__ _Float16 Xs[128][32];
  __shared__ _Float16 Ws[64][32];

  const int tid  = threadIdx.x;
  const int lane = tid & 31, wid = tid >> 5;
  const int waveM = (wid & 3) * 32, waveN = (wid >> 2) * 32;
  const int m0 = blockIdx.x * 128;
  const int n0 = blockIdx.y * 64;
  const int hi = lane >> 4, ln = lane & 15;

  v8f acc[2][2] = {};

  for (int k0 = 0; k0 < 256; k0 += 32) {
#pragma unroll
    for (int cc = 0; cc < 2; ++cc) {
      const int chunk = tid + cc * 256;
      const int row = chunk >> 2, kc = (chunk & 3) * 8;
      async_ld16(lds_off(&Xs[row][kc]),
                 xT + (size_t)(m0 + row) * 256 + k0 + kc);
    }
    {
      const int row = tid >> 2, kc = (tid & 3) * 8;
      async_ld16(lds_off(&Ws[row][kc]),
                 wh + (size_t)(n0 + row) * 256 + k0 + kc);
    }
    if (k0 + 32 < 256)
      __builtin_prefetch(xT + (size_t)(m0 + (tid >> 1)) * 256 + k0 + 32, 0, 1);
    wait_async0();
    __syncthreads();

#pragma unroll
    for (int ni = 0; ni < 2; ++ni) {
      const int col = waveN + ni * 16 + ln;
      v16h bf = *(const v16h*)&Ws[col][hi * 16];
#pragma unroll
      for (int mi = 0; mi < 2; ++mi) {
        const int row = waveM + mi * 16 + ln;
        v8h lo = *(const v8h*)&Xs[row][hi * 16];
        v8h up = *(const v8h*)&Xs[row][hi * 16 + 8];
        v16h af;
#pragma unroll
        for (int j = 0; j < 8; ++j) { af[j] = lo[j]; af[8 + j] = up[j]; }
        acc[mi][ni] = WMMA_F16(af, bf, acc[mi][ni]);
      }
    }
    __syncthreads();
  }

#pragma unroll
  for (int mi = 0; mi < 2; ++mi)
#pragma unroll
    for (int ni = 0; ni < 2; ++ni) {
      const int oc = n0 + waveN + ni * 16 + ln;
#pragma unroll
      for (int r = 0; r < 8; ++r) {
        const int m = m0 + waveM + mi * 16 + r + 8 * hi;
        const int bb = m >> 14, hw = m & (HWp - 1);
        outp[((size_t)(bb * 256 + oc)) * HWp + hw] = acc[mi][ni][r];
      }
    }
}

// ---------------------------------------------------------------------------
extern "C" void kernel_launch(void* const* d_in, const int* in_sizes, int n_in,
                              void* d_out, int out_size, void* d_ws,
                              size_t ws_size, hipStream_t stream) {
  (void)in_sizes; (void)n_in; (void)out_size; (void)ws_size;

  const float* x    = (const float*)d_in[0];
  const float* y    = (const float*)d_in[1];
  const float* qkvw = (const float*)d_in[2];
  const float* l1w  = (const float*)d_in[3];
  const float* l1s  = (const float*)d_in[4];
  const float* l1b  = (const float*)d_in[5];
  const float* l2w  = (const float*)d_in[6];
  const float* l2s  = (const float*)d_in[7];
  const float* l2b  = (const float*)d_in[8];
  const float* dww  = (const float*)d_in[9];
  const float* pbs  = (const float*)d_in[10];
  const float* pbb  = (const float*)d_in[11];
  const float* pww  = (const float*)d_in[12];
  const float* relb = (const float*)d_in[13];

  char* ws = (char*)d_ws;
  const size_t SZ_QKV = (size_t)768 * NPIX * sizeof(_Float16);  // 96 MB
  const size_t SZ_F32 = (size_t)256 * NPIX * sizeof(float);     // 64 MB
  const size_t SZ_XT  = (size_t)256 * NPIX * sizeof(_Float16);  // 32 MB

  _Float16* qkvT   = (_Float16*)ws;                      // [p][768] f16
  float*    osum   = (float*)ws;                         // aliases qkvT
  float*    attno  = (float*)(ws + SZ_QKV);
  _Float16* dwoutT = (_Float16*)(ws + SZ_QKV);           // aliases attno (dead)
  float*    localo = (float*)(ws + SZ_QKV + SZ_F32);
  float*    dwout  = (float*)(ws + SZ_QKV + 2 * SZ_F32);
  _Float16* xT     = (_Float16*)(ws + SZ_QKV + 3 * SZ_F32);
  _Float16* yT     = (_Float16*)(ws + SZ_QKV + 3 * SZ_F32 + SZ_XT);
  _Float16* w1h    = (_Float16*)(ws + SZ_QKV + 3 * SZ_F32 + 2 * SZ_XT);
  _Float16* qkvwh  = w1h + (size_t)10 * 256 * 256;
  _Float16* pwwh   = qkvwh + (size_t)768 * 256;

  // 0) one-time converts (f16 transposed activations, scale-folded weights)
  k_cvtw<<<(10 * 256 * 256 + 255) / 256, 256, 0, stream>>>(
      qkvw, l1w, l1s, l2w, l2s, pww, qkvwh, w1h, pwwh);
  k_cvtT<<<dim3(NPIX / 128, 256 / 32), 256, 0, stream>>>(x, xT);
  k_cvtT<<<dim3(NPIX / 128, 256 / 32), 256, 0, stream>>>(y, yT);

  // 1) qkv projection GEMM (WMMA, async-LDS feeds)
  k_qkv<<<dim3(NPIX / 128, 768 / 64), 256, 0, stream>>>(xT, qkvwh, qkvT);
  // 2) local branch implicit GEMM (WMMA, async-LDS feeds)
  k_local<<<dim3(4 * 128, 256 / 64), 256, 0, stream>>>(yT, w1h, l1b, l2b,
                                                       localo);
  // 3) window attention (WMMA)
  k_attn<<<1024 * 16, 32, 0, stream>>>(qkvT, relb, attno);
  // 4) pools + local sum (qkvT dead; osum reuses its space)
  k_pool<<<NPIX * 256 / 256, 256, 0, stream>>>(attno, localo, osum);
  // 5) depthwise 8x8 + BN (attno dead after this consumes osum)
  k_dw<<<NPIX * 256 / 256, 256, 0, stream>>>(osum, dww, pbs, pbb, dwout);
  // 6) transpose-convert dw output for the final async GEMM
  k_cvtT<<<dim3(NPIX / 128, 256 / 32), 256, 0, stream>>>(dwout, dwoutT);
  // 7) projection pointwise GEMM (WMMA, async-LDS feeds) -> final output
  k_pwa<<<dim3(NPIX / 128, 256 / 64), 256, 0, stream>>>(dwoutT, pwwh,
                                                        (float*)d_out);
}
